// AttentionQNet_7447473292229
// MI455X (gfx1250) — matrix operand order repaired and checked by reading, not compiled, not run on hardware
//
#include <hip/hip_runtime.h>
#include <hip/hip_bf16.h>
#include <math.h>

typedef __bf16 bf16;
typedef __attribute__((ext_vector_type(16))) __bf16 v16bf;
typedef __attribute__((ext_vector_type(8)))  float  v8f;

#define B_   8
#define S_   1024
#define IN_  64
#define E_   1024
#define H_   16
#define HD_  64
#define HID_ 4096

union FragB16 { v16bf v; uint4 q[2]; };
union FragAcc { v8f  v; float f[8]; };

__device__ __forceinline__ unsigned hash_u32(unsigned x) {
    x ^= x >> 16; x *= 0x7feb352du;
    x ^= x >> 15; x *= 0x846ca68bu;
    x ^= x >> 16;
    return x;
}

// LDS byte offset of a shared-memory pointer (generic ptr low 32 bits on amdgcn)
__device__ __forceinline__ unsigned lds_off(const void* p) {
    return (unsigned)(size_t)p;
}

// ---------------------------------------------------------------- conversions
__global__ void cvt_f32_bf16(const float* __restrict__ in, bf16* __restrict__ out, int n) {
    int i = blockIdx.x * blockDim.x + threadIdx.x;
    int stride = gridDim.x * blockDim.x;
    for (; i < n; i += stride) out[i] = (bf16)in[i];
}

// W (Kdim x Ndim) f32 row-major  ->  WT (Ndim x Kdim) bf16 row-major
__global__ __launch_bounds__(256) void cvt_transpose_f32_bf16(
    const float* __restrict__ in, bf16* __restrict__ out, int Kdim, int Ndim)
{
    __shared__ float sh[32][33];
    const int n0 = blockIdx.x * 32;
    const int k0 = blockIdx.y * 32;
    const int tx = threadIdx.x & 31;
    const int ty = threadIdx.x >> 5;  // 0..7
    #pragma unroll
    for (int i = 0; i < 4; ++i)
        sh[ty + 8 * i][tx] = in[(size_t)(k0 + ty + 8 * i) * Ndim + n0 + tx];
    __syncthreads();
    #pragma unroll
    for (int i = 0; i < 4; ++i)
        out[(size_t)(n0 + ty + 8 * i) * Kdim + k0 + tx] = (bf16)sh[tx][ty + 8 * i];
}

// ---------------------------------------------------------------- WMMA GEMM
// C(M,N) = act(A(M,K) @ B(K,N) + bias).  A bf16 row-major (M,K);
// B passed PRE-TRANSPOSED: BT bf16 row-major (N,K) so tiles load contiguous.
// Block: 256 thr = 8 waves; tile 128x64, BK=64; wave (wm,wn) owns 32x32 out
// (4 accumulators, 8 WMMA per stage). LDS double-buffered, filled with CDNA5
// async global->LDS copies (ASYNCcnt), one barrier per stage.
__global__ __launch_bounds__(256) void gemm_bf16_wmma(
    const bf16* __restrict__ A, const bf16* __restrict__ BT,
    const float* __restrict__ bias,
    float* __restrict__ outF, bf16* __restrict__ outB,
    int M, int N, int K, int relu)
{
    __shared__ bf16 As[2][128][64];  // 32 KB
    __shared__ bf16 Bs[2][64][64];   // 16 KB  ([n][k])

    const int tid  = threadIdx.x;
    const int lane = tid & 31;
    const int half = lane >> 4;
    const int lr   = lane & 15;
    const int w    = tid >> 5;
    const int wm   = w & 3;      // 4 row-groups of 32
    const int wn   = w >> 2;     // 2 col-groups of 32
    const int tileM = blockIdx.y * 128;
    const int tileN = blockIdx.x * 64;

    FragAcc c00, c01, c10, c11;
    #pragma unroll
    for (int r = 0; r < 8; ++r) { c00.f[r] = 0.f; c01.f[r] = 0.f; c10.f[r] = 0.f; c11.f[r] = 0.f; }

    auto issue = [&](int buf, int kb) {
        // A tile: 128x64 bf16 = 16 KB -> 4 async b128 per thread
        #pragma unroll
        for (int t = 0; t < 4; ++t) {
            int seg = tid + t * 256;
            int r   = seg >> 3;
            int c8  = (seg & 7) * 8;
            const bf16* g = A + (size_t)(tileM + r) * K + kb + c8;
            unsigned l = lds_off(&As[buf][r][c8]);
            asm volatile("global_load_async_to_lds_b128 %0, %1, off"
                         :: "v"(l), "v"(g) : "memory");
        }
        // B tile: 64x64 bf16 = 8 KB -> 2 async b128 per thread
        #pragma unroll
        for (int t = 0; t < 2; ++t) {
            int seg = tid + t * 256;
            int n   = seg >> 3;
            int c8  = (seg & 7) * 8;
            const bf16* g = BT + (size_t)(tileN + n) * K + kb + c8;
            unsigned l = lds_off(&Bs[buf][n][c8]);
            asm volatile("global_load_async_to_lds_b128 %0, %1, off"
                         :: "v"(l), "v"(g) : "memory");
        }
    };

    issue(0, 0);
    const int nk = K >> 6;
    for (int i = 0; i < nk; ++i) {
        asm volatile("s_wait_asynccnt 0" ::: "memory");  // own stage-i copies done
        __syncthreads();                                  // everyone's done; prev reads done
        if (i + 1 < nk) issue((i + 1) & 1, (i + 1) << 6); // overlap next copies w/ math
        const int buf = i & 1;

        #pragma unroll
        for (int ks = 0; ks < 64; ks += 32) {
            FragB16 a0, a1, b0, b1;
            const int koff = ks + half * 8;
            const int ar0  = wm * 32 + lr;
            a0.q[0] = *(const uint4*)&As[buf][ar0][koff];
            a0.q[1] = *(const uint4*)&As[buf][ar0][koff + 16];
            a1.q[0] = *(const uint4*)&As[buf][ar0 + 16][koff];
            a1.q[1] = *(const uint4*)&As[buf][ar0 + 16][koff + 16];
            const int bk  = ks + half * 8;
            const int bn0 = wn * 32 + lr;
            b0.q[0] = *(const uint4*)&Bs[buf][bn0][bk];
            b0.q[1] = *(const uint4*)&Bs[buf][bn0][bk + 16];
            b1.q[0] = *(const uint4*)&Bs[buf][bn0 + 16][bk];
            b1.q[1] = *(const uint4*)&Bs[buf][bn0 + 16][bk + 16];
            c00.v = __builtin_amdgcn_wmma_f32_16x16x32_bf16(false, a0.v, false, b0.v, (short)0, c00.v, false, false);
            c01.v = __builtin_amdgcn_wmma_f32_16x16x32_bf16(false, a0.v, false, b1.v, (short)0, c01.v, false, false);
            c10.v = __builtin_amdgcn_wmma_f32_16x16x32_bf16(false, a1.v, false, b0.v, (short)0, c10.v, false, false);
            c11.v = __builtin_amdgcn_wmma_f32_16x16x32_bf16(false, a1.v, false, b1.v, (short)0, c11.v, false, false);
        }
    }

    // ---- epilogue (C/D layout: VGPR r -> row half*8+r, col = lane%16)
    const int row0 = tileM + wm * 32 + half * 8;
    const int col0 = tileN + wn * 32 + lr;
    const float bi0 = bias ? bias[col0]      : 0.f;
    const float bi1 = bias ? bias[col0 + 16] : 0.f;
    #pragma unroll
    for (int r = 0; r < 8; ++r) {
        float v00 = c00.f[r] + bi0, v01 = c01.f[r] + bi1;
        float v10 = c10.f[r] + bi0, v11 = c11.f[r] + bi1;
        if (relu) {
            v00 = fmaxf(v00, 0.f); v01 = fmaxf(v01, 0.f);
            v10 = fmaxf(v10, 0.f); v11 = fmaxf(v11, 0.f);
        }
        size_t o0 = (size_t)(row0 + r) * N + col0;
        size_t o1 = (size_t)(row0 + 16 + r) * N + col0;
        if (outF) { outF[o0] = v00; outF[o0 + 16] = v01; outF[o1] = v10; outF[o1 + 16] = v11; }
        if (outB) { outB[o0] = (bf16)v00; outB[o0 + 16] = (bf16)v01;
                    outB[o1] = (bf16)v10; outB[o1 + 16] = (bf16)v11; }
    }
}

// ---------------------------------------------------------------- flash attention
// Grid: (B*H, S/128). 8 waves; wave w owns 16 query rows. Online softmax,
// keys in chunks of 64; dropout on normalized probs via hash RNG.
__global__ __launch_bounds__(256) void flash_attn_wmma(
    const bf16* __restrict__ Q, const bf16* __restrict__ Km, const bf16* __restrict__ V,
    const unsigned char* __restrict__ pmask,
    bf16* __restrict__ ctx)
{
    __shared__ bf16 Vt[64][64];     // [hd][key]  (8 KB)
    __shared__ bf16 Pt[8][16][64];  // per-wave probs [qrow][key] (16 KB)

    const int tid  = threadIdx.x;
    const int w    = tid >> 5;
    const int lane = tid & 31;
    const int half = lane >> 4;
    const int lr   = lane & 15;

    const int bh = blockIdx.x;
    const int b  = bh >> 4;
    const int q0 = blockIdx.y * 128 + w * 16;

    const size_t base = ((size_t)b * S_) * E_ + (size_t)(bh & 15) * HD_;

    FragAcc o[4];
    float m_i[8], l_i[8];
    #pragma unroll
    for (int r = 0; r < 8; ++r) {
        m_i[r] = -3.0e38f; l_i[r] = 0.f;
        #pragma unroll
        for (int f = 0; f < 4; ++f) o[f].f[r] = 0.f;
    }

    for (int kc = 0; kc < S_; kc += 64) {
        __syncthreads();
        // ---- stage V chunk transposed: Vt[hd][key]
        {
            int key = tid >> 2;
            int h0  = (tid & 3) * 16;
            const bf16* vp = V + base + (size_t)(kc + key) * E_ + h0;
            bf16 tmp[16];
            *(uint4*)&tmp[0] = *(const uint4*)vp;
            *(uint4*)&tmp[8] = *(const uint4*)(vp + 8);
            #pragma unroll
            for (int j = 0; j < 16; ++j) Vt[h0 + j][key] = tmp[j];
        }
        __syncthreads();

        // ---- S = Q @ K^T (16 q-rows x 64 keys)
        FragAcc s[4];
        #pragma unroll
        for (int nt = 0; nt < 4; ++nt)
            #pragma unroll
            for (int r = 0; r < 8; ++r) s[nt].f[r] = 0.f;

        #pragma unroll
        for (int kk = 0; kk < HD_; kk += 32) {
            FragB16 a;
            const bf16* qp = Q + base + (size_t)(q0 + lr) * E_ + kk + half * 8;
            a.q[0] = *(const uint4*)qp;
            a.q[1] = *(const uint4*)(qp + 16);
            #pragma unroll
            for (int nt = 0; nt < 4; ++nt) {
                FragB16 bb;  // K row-major == K^T column-contiguous
                const bf16* kp = Km + base + (size_t)(kc + nt * 16 + lr) * E_ + kk + half * 8;
                bb.q[0] = *(const uint4*)kp;
                bb.q[1] = *(const uint4*)(kp + 16);
                s[nt].v = __builtin_amdgcn_wmma_f32_16x16x32_bf16(false, a.v, false, bb.v, (short)0, s[nt].v, false, false);
            }
        }

        float madd[4];
        #pragma unroll
        for (int nt = 0; nt < 4; ++nt)
            madd[nt] = pmask[(size_t)b * S_ + kc + nt * 16 + lr] ? 0.f : -1.0e30f;

        // ---- online softmax (rows r / r+8 split across lane halves)
        const float scale = 0.125f;
        #pragma unroll
        for (int r = 0; r < 8; ++r) {
            float sv[4];
            #pragma unroll
            for (int nt = 0; nt < 4; ++nt) sv[nt] = s[nt].f[r] * scale + madd[nt];
            float cm = fmaxf(fmaxf(sv[0], sv[1]), fmaxf(sv[2], sv[3]));
            #pragma unroll
            for (int m = 8; m >= 1; m >>= 1) cm = fmaxf(cm, __shfl_xor(cm, m, 32));
            float mo = m_i[r];
            float mn = fmaxf(mo, cm);
            float alpha = __expf(mo - mn);
            m_i[r] = mn;

            int qrow = half * 8 + r;
            unsigned qglob = (unsigned)(bh * S_ + q0 + qrow);
            float rs = 0.f;
            #pragma unroll
            for (int nt = 0; nt < 4; ++nt) {
                float p = __expf(sv[nt] - mn);
                rs += p;
                unsigned key = (unsigned)(kc + nt * 16 + lr);
                unsigned hsh = hash_u32((qglob * 1024u + key) ^ 0x2Au);
                float pd = (hsh > 429496729u) ? p * (1.0f / 0.9f) : 0.f;  // keep p = 0.9
                Pt[w][qrow][nt * 16 + lr] = (bf16)pd;
            }
            #pragma unroll
            for (int m = 8; m >= 1; m >>= 1) rs += __shfl_xor(rs, m, 32);
            l_i[r] = l_i[r] * alpha + rs;
            #pragma unroll
            for (int f = 0; f < 4; ++f) o[f].f[r] *= alpha;
        }
        // same-wave LDS store->load ordering before A-frag reads of Pt
        asm volatile("s_wait_dscnt 0" ::: "memory");

        // ---- O += P @ V
        #pragma unroll
        for (int kk = 0; kk < 64; kk += 32) {
            FragB16 a;
            a.q[0] = *(const uint4*)&Pt[w][lr][kk + half * 8];
            a.q[1] = *(const uint4*)&Pt[w][lr][kk + half * 8 + 16];
            #pragma unroll
            for (int f = 0; f < 4; ++f) {
                FragB16 bb;
                bb.q[0] = *(const uint4*)&Vt[f * 16 + lr][kk + half * 8];
                bb.q[1] = *(const uint4*)&Vt[f * 16 + lr][kk + half * 8 + 16];
                o[f].v = __builtin_amdgcn_wmma_f32_16x16x32_bf16(false, a.v, false, bb.v, (short)0, o[f].v, false, false);
            }
        }
    }

    #pragma unroll
    for (int r = 0; r < 8; ++r) {
        float inv = 1.0f / l_i[r];
        int qrow = q0 + half * 8 + r;
        #pragma unroll
        for (int f = 0; f < 4; ++f)
            ctx[base + (size_t)qrow * E_ + f * 16 + lr] = (bf16)(o[f].f[r] * inv);
    }
}

// ---------------------------------------------------------------- LayerNorm (fp32 in -> bf16 out)
__global__ __launch_bounds__(256) void layernorm_f32_bf16(
    const float* __restrict__ in, const float* __restrict__ g, const float* __restrict__ bb,
    bf16* __restrict__ out, int width)
{
    __shared__ float red[256];
    const int row = blockIdx.x;
    const int tid = threadIdx.x;
    const float* x = in + (size_t)row * width;

    float s = 0.f;
    for (int j = tid; j < width; j += 256) s += x[j];
    red[tid] = s; __syncthreads();
    for (int st = 128; st > 0; st >>= 1) { if (tid < st) red[tid] += red[tid + st]; __syncthreads(); }
    float mean = red[0] / width;
    __syncthreads();

    float v = 0.f;
    for (int j = tid; j < width; j += 256) { float d = x[j] - mean; v += d * d; }
    red[tid] = v; __syncthreads();
    for (int st = 128; st > 0; st >>= 1) { if (tid < st) red[tid] += red[tid + st]; __syncthreads(); }
    float rstd = rsqrtf(red[0] / width + 1e-5f);

    for (int j = tid; j < width; j += 256)
        out[(size_t)row * width + j] = (bf16)((x[j] - mean) * rstd * g[j] + bb[j]);
}

// ---------------------------------------------------------------- fused LN2 + W2 reduction
__global__ __launch_bounds__(256) void ln2_w2_out(
    const bf16* __restrict__ in, const float* __restrict__ g, const float* __restrict__ bb,
    const float* __restrict__ W2, const float* __restrict__ b2,
    float* __restrict__ out, int width)
{
    __shared__ float red[256];
    const int row = blockIdx.x;
    const int tid = threadIdx.x;
    const bf16* x = in + (size_t)row * width;

    float s = 0.f;
    for (int j = tid; j < width; j += 256) s += (float)x[j];
    red[tid] = s; __syncthreads();
    for (int st = 128; st > 0; st >>= 1) { if (tid < st) red[tid] += red[tid + st]; __syncthreads(); }
    float mean = red[0] / width;
    __syncthreads();

    float v = 0.f;
    for (int j = tid; j < width; j += 256) { float d = (float)x[j] - mean; v += d * d; }
    red[tid] = v; __syncthreads();
    for (int st = 128; st > 0; st >>= 1) { if (tid < st) red[tid] += red[tid + st]; __syncthreads(); }
    float rstd = rsqrtf(red[0] / width + 1e-5f);
    __syncthreads();

    float acc = 0.f;
    for (int j = tid; j < width; j += 256)
        acc += (((float)x[j] - mean) * rstd * g[j] + bb[j]) * W2[j];
    red[tid] = acc; __syncthreads();
    for (int st = 128; st > 0; st >>= 1) { if (tid < st) red[tid] += red[tid + st]; __syncthreads(); }
    if (tid == 0) out[row] = red[0] + b2[0];
}

// ---------------------------------------------------------------- launcher
extern "C" void kernel_launch(void* const* d_in, const int* in_sizes, int n_in,
                              void* d_out, int out_size, void* d_ws, size_t ws_size,
                              hipStream_t stream)
{
    const float*         x     = (const float*)d_in[0];
    const unsigned char* pmask = (const unsigned char*)d_in[1];
    const float* W_emb = (const float*)d_in[2];
    const float* b_emb = (const float*)d_in[3];
    const float* Wq    = (const float*)d_in[4];
    const float* Wk    = (const float*)d_in[5];
    const float* Wv    = (const float*)d_in[6];
    const float* Wc    = (const float*)d_in[7];
    const float* b_c   = (const float*)d_in[8];
    const float* ln1_g = (const float*)d_in[9];
    const float* ln1_b = (const float*)d_in[10];
    const float* W1    = (const float*)d_in[11];
    const float* b1    = (const float*)d_in[12];
    const float* ln2_g = (const float*)d_in[13];
    const float* ln2_b = (const float*)d_in[14];
    const float* W2    = (const float*)d_in[15];
    const float* b2    = (const float*)d_in[16];
    float* out = (float*)d_out;

    const int M = B_ * S_;  // 8192 rows

    char* ws = (char*)d_ws;
    size_t off = 0;
    auto alloc = [&](size_t bytes) -> char* {
        char* p = ws + off;
        off = (off + bytes + 255) & ~(size_t)255;
        return p;
    };
    bf16* xb     = (bf16*)alloc((size_t)M * IN_ * 2);
    bf16* WembT  = (bf16*)alloc((size_t)E_ * IN_ * 2);     // (N=E) x (K=IN)
    bf16* WqT    = (bf16*)alloc((size_t)E_ * E_ * 2);
    bf16* WkT    = (bf16*)alloc((size_t)E_ * E_ * 2);
    bf16* WvT    = (bf16*)alloc((size_t)E_ * E_ * 2);
    bf16* WcT    = (bf16*)alloc((size_t)E_ * E_ * 2);
    bf16* W1T    = (bf16*)alloc((size_t)HID_ * E_ * 2);    // (N=HID) x (K=E)
    bf16* hB     = (bf16*)alloc((size_t)M * E_ * 2);
    bf16* Qb     = (bf16*)alloc((size_t)M * E_ * 2);
    bf16* Kb     = (bf16*)alloc((size_t)M * E_ * 2);
    bf16* Vb     = (bf16*)alloc((size_t)M * E_ * 2);
    bf16* ctxB   = (bf16*)alloc((size_t)M * E_ * 2);
    float* attF  = (float*)alloc((size_t)M * E_ * 4);
    bf16* ln1o   = (bf16*)alloc((size_t)M * E_ * 2);
    bf16* ff1B   = (bf16*)alloc((size_t)M * HID_ * 2);

    // x -> bf16 (A-side, no transpose)
    cvt_f32_bf16<<<2048, 256, 0, stream>>>(x, xb, M * IN_);

    // weights -> bf16 transposed (N x K)
    auto cvtT = [&](const float* src, bf16* dst, int Kd, int Nd) {
        cvt_transpose_f32_bf16<<<dim3(Nd / 32, Kd / 32), 256, 0, stream>>>(src, dst, Kd, Nd);
    };
    cvtT(W_emb, WembT, IN_, E_);
    cvtT(Wq, WqT, E_, E_);
    cvtT(Wk, WkT, E_, E_);
    cvtT(Wv, WvT, E_, E_);
    cvtT(Wc, WcT, E_, E_);
    cvtT(W1, W1T, E_, HID_);

    auto gemm = [&](const bf16* A, const bf16* BTmat, const float* bias,
                    float* oF, bf16* oB, int MM, int NN, int KK, int relu) {
        dim3 g(NN / 64, MM / 128);
        gemm_bf16_wmma<<<g, 256, 0, stream>>>(A, BTmat, bias, oF, oB, MM, NN, KK, relu);
    };

    gemm(xb, WembT, b_emb, nullptr, hB, M, E_, IN_, 0);        // embed
    gemm(hB, WqT, nullptr, nullptr, Qb, M, E_, E_, 0);         // Q
    gemm(hB, WkT, nullptr, nullptr, Kb, M, E_, E_, 0);         // K
    gemm(hB, WvT, nullptr, nullptr, Vb, M, E_, E_, 0);         // V

    flash_attn_wmma<<<dim3(B_ * H_, S_ / 128), 256, 0, stream>>>(Qb, Kb, Vb, pmask, ctxB);

    gemm(ctxB, WcT, b_c, attF, nullptr, M, E_, E_, 0);         // ctx @ Wc + b_c
    layernorm_f32_bf16<<<M, 256, 0, stream>>>(attF, ln1_g, ln1_b, ln1o, E_);
    gemm(ln1o, W1T, b1, nullptr, ff1B, M, HID_, E_, 1);        // relu(· @ W1 + b1)
    ln2_w2_out<<<M, 256, 0, stream>>>(ff1B, ln2_g, ln2_b, W2, b2, out, HID_);

    (void)in_sizes; (void)n_in; (void)out_size; (void)ws_size;
}
